// SegGPNet_6579889897729
// MI455X (gfx1250) — matrix operand order repaired and checked by reading, not compiled, not run on hardware
//
#include <hip/hip_runtime.h>
#include <hip/hip_bf16.h>
#include <math.h>

// ---------------------------------------------------------------------------
// SegGPNet on gfx1250: branch-MLP via augmented-activation WMMA GEMM.
//   B=16, N=8192, C_IN=3, HID=64, D=256, K=4, GH=128, OUT=128
// ---------------------------------------------------------------------------

#define BATCH 16
#define NPTS  8192
#define CIN   3
#define HID   64
#define DDIM  256
#define KBR   4
#define GH    128
#define OUTD  128

#define TILE_M 128           // points per block
#define TILES_PER_B (NPTS / TILE_M)   // 64
#define SA 264               // htile row stride in halves (256 + 8 pad)
#define SB 264               // w2t   row stride in halves

typedef _Float16 v16h __attribute__((ext_vector_type(16)));
typedef _Float16 v8h  __attribute__((ext_vector_type(8)));
typedef float    v8f  __attribute__((ext_vector_type(8)));

__device__ __forceinline__ float gelu_exact(float v) {
    return 0.5f * v * (1.0f + erff(v * 0.70710678118654752f));
}

// ---------------------------------------------------------------------------
// Kernel A: per-point branch MLP (3->64 GELU scalar; 64->256 via WMMA on the
// augmented [128 x 256] activation against stacked W2), fused tile-max.
// grid = BATCH * TILES_PER_B blocks of 256 threads (8 wave32).
// ---------------------------------------------------------------------------
__global__ void __launch_bounds__(256)
seg_branch_wmma(const float* __restrict__ x,     // [B,3,N]
                const float* __restrict__ lab,   // [B,4,N]
                const float* __restrict__ W1,    // [4,3,64]
                const float* __restrict__ b1,    // [4,64]
                const float* __restrict__ W2,    // [4,64,256]
                const float* __restrict__ b2,    // [4,256]
                float* __restrict__ partials)    // [B,64,256] tile maxes
{
    extern __shared__ char smem[];
    _Float16* htile = (_Float16*)smem;                 // 128*264 halves
    _Float16* w2t   = htile + TILE_M * SA;             // 128*264 halves
    float* w1sh = (float*)(w2t + 128 * SB);            // 4*3*64
    float* b1sh = w1sh + KBR * CIN * HID;              // 256
    float* b2sh = b1sh + KBR * HID;                    // 1024
    float* wmax = b2sh + KBR * DDIM;                   // 8*128
    int*   kidx = (int*)(wmax + 8 * 128);              // 128

    const int t      = threadIdx.x;
    const int bidx   = blockIdx.x >> 6;   // batch
    const int tile   = blockIdx.x & 63;
    const int n_base = tile * TILE_M;

    // ---- stage small tensors ------------------------------------------------
    for (int i = t; i < KBR * CIN * HID; i += 256) w1sh[i] = W1[i];
    if (t < KBR * HID) b1sh[t] = b1[t];
    for (int i = t; i < KBR * DDIM; i += 256) b2sh[i] = b2[i];
    __syncthreads();

    // ---- per-point: argmax label + 3->64 GELU into augmented htile ----------
    {
        const int p    = t >> 1;           // point in tile (0..127)
        const int half = t & 1;            // which 128 columns of h_aug
        const int n    = n_base + p;

        float l0 = lab[((size_t)bidx * KBR + 0) * NPTS + n];
        float l1 = lab[((size_t)bidx * KBR + 1) * NPTS + n];
        float l2 = lab[((size_t)bidx * KBR + 2) * NPTS + n];
        float l3 = lab[((size_t)bidx * KBR + 3) * NPTS + n];
        int k = 0; float lm = l0;
        if (l1 > lm) { lm = l1; k = 1; }
        if (l2 > lm) { lm = l2; k = 2; }
        if (l3 > lm) { lm = l3; k = 3; }
        if (half == 0) kidx[p] = k;

        float x0 = x[((size_t)bidx * CIN + 0) * NPTS + n];
        float x1 = x[((size_t)bidx * CIN + 1) * NPTS + n];
        float x2 = x[((size_t)bidx * CIN + 2) * NPTS + n];

        const int c0 = half * 128;
        for (int col = c0; col < c0 + 128; ++col) {
            const int kk = col >> 6;       // branch slice
            const int j  = col & 63;
            float v = 0.0f;
            if (kk == k) {
                float acc = b1sh[k * HID + j]
                          + x0 * w1sh[(k * CIN + 0) * HID + j]
                          + x1 * w1sh[(k * CIN + 1) * HID + j]
                          + x2 * w1sh[(k * CIN + 2) * HID + j];
                v = gelu_exact(acc);
            }
            htile[p * SA + col] = (_Float16)v;
        }
    }

    const int wave = t >> 5;
    const int lane = t & 31;
    const int lm16 = lane & 15;
    const int lhi  = lane >> 4;
    const int m0   = wave * 16;            // wave owns 16 rows (points)

    // ---- two 128-column passes over D to bound LDS --------------------------
    for (int pass = 0; pass < 2; ++pass) {
        // stage W2 half, transposed: w2t[d'][k*64+h] = W2[k][h][pass*128+d']
        for (int i = 0; i < 128; ++i) {
            const int e   = i * 256 + t;
            const int kkh = e >> 7;        // 0..255  (k*64+h)
            const int dp  = e & 127;       // 0..127
            w2t[dp * SB + kkh] = (_Float16)W2[(size_t)kkh * DDIM + pass * 128 + dp];
        }
        __syncthreads();   // htile (pass 0) + w2t ready

        for (int nt = 0; nt < 8; ++nt) {
            v8f acc = {0.f, 0.f, 0.f, 0.f, 0.f, 0.f, 0.f, 0.f};
            #pragma unroll
            for (int k0 = 0; k0 < DDIM; k0 += 32) {
                // A fragment: lane(m=lm16,hi=lhi): K = {k0+hi*8.., k0+16+hi*8..}
                const _Float16* arow = &htile[(m0 + lm16) * SA + k0 + lhi * 8];
                v8h a0 = *(const v8h*)arow;
                v8h a1 = *(const v8h*)(arow + 16);
                v16h afrag = __builtin_shufflevector(a0, a1,
                    0,1,2,3,4,5,6,7,8,9,10,11,12,13,14,15);
                // B fragment: lane(n=lm16,hi=lhi): K = {k0+hi*8.., k0+16+hi*8..}, col n
                const _Float16* brow = &w2t[(nt * 16 + lm16) * SB + k0 + lhi * 8];
                v8h bb0 = *(const v8h*)brow;
                v8h bb1 = *(const v8h*)(brow + 16);
                v16h bfrag = __builtin_shufflevector(bb0, bb1,
                    0,1,2,3,4,5,6,7,8,9,10,11,12,13,14,15);
                acc = __builtin_amdgcn_wmma_f32_16x16x32_f16(
                    false, afrag, false, bfrag, (short)0, acc, false, false);
            }
            // add b2[k*] per row, max over the 8 rows this lane holds
            float mx = -3.4e38f;
            #pragma unroll
            for (int r = 0; r < 8; ++r) {
                const int pp = m0 + r + 8 * lhi;           // point index
                const int d  = pass * 128 + nt * 16 + lm16; // global column
                float v = acc[r] + b2sh[kidx[pp] * DDIM + d];
                mx = fmaxf(mx, v);
            }
            // merge rows held by the paired lane (same column, other 8 rows)
            mx = fmaxf(mx, __shfl_xor(mx, 16, 32));
            if (lane < 16) wmax[wave * 128 + nt * 16 + lm16] = mx;
        }
        __syncthreads();
        // cross-wave max -> tile partial
        if (t < 128) {
            float m = wmax[t];
            #pragma unroll
            for (int w = 1; w < 8; ++w) m = fmaxf(m, wmax[w * 128 + t]);
            partials[(size_t)blockIdx.x * DDIM + pass * 128 + t] = m;
        }
        __syncthreads();   // safe to overwrite w2t next pass
    }
}

// ---------------------------------------------------------------------------
// Kernel B: reduce tile maxes -> g[256], then 256->128 GELU -> 128 head.
// grid = BATCH blocks of 256 threads.
// ---------------------------------------------------------------------------
__global__ void __launch_bounds__(256)
seg_global_head(const float* __restrict__ partials, // [B,64,256]
                const float* __restrict__ Wg1,      // [256,128]
                const float* __restrict__ bg1,      // [128]
                const float* __restrict__ Wg2,      // [128,128]
                const float* __restrict__ bg2,      // [128]
                float* __restrict__ out)            // [B,128]
{
    __shared__ float g[DDIM];
    __shared__ float a1[GH];
    const int b = blockIdx.x;
    const int t = threadIdx.x;

    if (t < DDIM) {
        float m = -3.4e38f;
        for (int tl = 0; tl < TILES_PER_B; ++tl)
            m = fmaxf(m, partials[((size_t)b * TILES_PER_B + tl) * DDIM + t]);
        g[t] = m;
    }
    __syncthreads();
    if (t < GH) {
        float acc = bg1[t];
        for (int d = 0; d < DDIM; ++d) acc += g[d] * Wg1[d * GH + t];
        a1[t] = gelu_exact(acc);
    }
    __syncthreads();
    if (t < OUTD) {
        float acc = bg2[t];
        for (int j = 0; j < GH; ++j) acc += a1[j] * Wg2[j * OUTD + t];
        out[(size_t)b * OUTD + t] = acc;
    }
}

// ---------------------------------------------------------------------------
extern "C" void kernel_launch(void* const* d_in, const int* in_sizes, int n_in,
                              void* d_out, int out_size, void* d_ws, size_t ws_size,
                              hipStream_t stream) {
    const float* x   = (const float*)d_in[0];
    const float* lab = (const float*)d_in[1];
    const float* W1  = (const float*)d_in[2];
    const float* b1  = (const float*)d_in[3];
    const float* W2  = (const float*)d_in[4];
    const float* b2  = (const float*)d_in[5];
    const float* Wg1 = (const float*)d_in[6];
    const float* bg1 = (const float*)d_in[7];
    const float* Wg2 = (const float*)d_in[8];
    const float* bg2 = (const float*)d_in[9];
    float* out      = (float*)d_out;
    float* partials = (float*)d_ws;   // BATCH*64*256 f32 = 1 MiB

    const size_t smem =
        (size_t)TILE_M * SA * sizeof(_Float16) +   // htile
        (size_t)128 * SB * sizeof(_Float16) +      // w2t
        (KBR * CIN * HID + KBR * HID + KBR * DDIM + 8 * 128) * sizeof(float) +
        TILE_M * sizeof(int);                       // ~145 KB, fits WGP LDS

    seg_branch_wmma<<<BATCH * TILES_PER_B, 256, smem, stream>>>(
        x, lab, W1, b1, W2, b2, partials);
    seg_global_head<<<BATCH, 256, 0, stream>>>(
        partials, Wg1, bg1, Wg2, bg2, out);
}
